// ViT_c_22574348108543
// MI455X (gfx1250) — compile-verified, matrix operand order checked by reading, not run on hardware
//
#include <hip/hip_runtime.h>
#include <hip/hip_bf16.h>
#include <math.h>

// ---------------- model constants ----------------
#define Lc 12
#define Hc 12
#define Dc 768
#define Nc 197
#define Mc 3072
#define Pc 16
#define Cc 3
#define IMGc 224
#define Bc 16

typedef __attribute__((ext_vector_type(16))) __bf16 v16bf;
typedef __attribute__((ext_vector_type(8)))  float  v8f;
typedef __attribute__((ext_vector_type(4)))  unsigned int u32x4;
typedef __attribute__((ext_vector_type(2)))  unsigned int u32x2;

union FragU { v16bf bf; u32x4 u[2]; };

__device__ inline unsigned int bfround(float f) {
    unsigned int u = __float_as_uint(f);
    return u + 0x7fffu + ((u >> 16) & 1u);   // RNE; result in high 16 bits
}
__device__ inline unsigned short f2bf(float f) {
    return (unsigned short)(bfround(f) >> 16);
}
// pack two floats -> two bf16 in one dword via v_perm_b32
// perm(a,b,sel): bytes 0-3 index into b, 4-7 into a
__device__ inline unsigned int pack2bf(float a, float b) {
    return __builtin_amdgcn_perm(bfround(b), bfround(a), 0x07060302u);
}

// ---------------- generic WMMA GEMM ----------------
// C[z][m][n] = sum_k A[zA][m][k] * B'[zB][k][n]  (+bias, opt. GELU)
// TRANSB=true : B stored [N,K] (weight style, C = A @ B^T)
// TRANSB=false: B stored [K,N]
// BIASMODE: 0 none, 1 per-column (n), 2 per-row (m)
static constexpr int BM = 128, BN = 128, BK = 32;
static constexpr int LDSK = BK + 8;   // ushort stride (80B rows, 16B aligned)

template<bool TRANSB, int BIASMODE, bool GELU>
__global__ __launch_bounds__(256)
void gemm_wmma(const float* __restrict__ A, const float* __restrict__ Bm,
               const float* __restrict__ bias, float* __restrict__ C,
               int Mdim, int Ndim, int Kdim, int lda, int ldb, int ldc,
               long long strideA, int zdivA,
               long long strideB, int zmodB,
               long long strideC, long long strideBias)
{
    __shared__ __align__(16) unsigned short sA[2][BM * LDSK];
    __shared__ __align__(16) unsigned short sB[2][BN * LDSK];

    const int z = blockIdx.z;
    A  += (long long)(z / zdivA) * strideA;
    Bm += (long long)(z % zmodB) * strideB;
    C  += (long long)z * strideC;
    const float* biasz = bias ? bias + (long long)(z % zmodB) * strideBias : nullptr;

    const int m0 = blockIdx.y * BM;
    const int n0 = blockIdx.x * BN;
    const int tid  = threadIdx.x;
    const int lane = tid & 31;
    const int wave = tid >> 5;
    const int wm = wave >> 2;   // 0..1  (64-row slab)
    const int wn = wave & 3;    // 0..3  (32-col slab)

    const bool aRowsFull = (m0 + BM <= Mdim) && ((lda & 3) == 0);
    const bool bColsFull = (n0 + BN <= Ndim) && ((ldb & 3) == 0);

    v8f acc[4][2];
    #pragma unroll
    for (int mt = 0; mt < 4; ++mt)
        #pragma unroll
        for (int nt = 0; nt < 2; ++nt)
            #pragma unroll
            for (int i = 0; i < 8; ++i) acc[mt][nt][i] = 0.0f;

    const int rsel = lane & 15;
    const int hsel = (lane >> 4) * 8;    // half-index offset (0 or 8)

    // ---- staged fetch: global -> registers (branch-free slow paths) ----
    float4 ra[4], rb[4];
    auto fetchA = [&](int k0) {
        if (aRowsFull && (k0 + BK <= Kdim)) {
            #pragma unroll
            for (int i = 0; i < 4; ++i) {
                int idx = tid + i * 256;
                int r = idx >> 3, c4 = (idx & 7) << 2;
                ra[i] = *(const float4*)(A + (long long)(m0 + r) * lda + k0 + c4);
            }
        } else {
            #pragma unroll
            for (int i = 0; i < 4; ++i) {
                int idx = tid + i * 256;
                int r = idx >> 3, c4 = (idx & 7) << 2;
                int gm = m0 + r;
                int gmc = min(gm, Mdim - 1);
                float4 v;
                #pragma unroll
                for (int j = 0; j < 4; ++j) {
                    int gk = k0 + c4 + j;
                    int gkc = min(gk, Kdim - 1);
                    float t = A[(long long)gmc * lda + gkc];     // clamped, always in-bounds
                    (&v.x)[j] = (gm < Mdim && gk < Kdim) ? t : 0.0f;
                }
                ra[i] = v;
            }
        }
    };
    auto fetchB = [&](int k0) {
        if (TRANSB) {
            if (bColsFull && (k0 + BK <= Kdim)) {
                #pragma unroll
                for (int i = 0; i < 4; ++i) {
                    int idx = tid + i * 256;
                    int r = idx >> 3, c4 = (idx & 7) << 2;
                    rb[i] = *(const float4*)(Bm + (long long)(n0 + r) * ldb + k0 + c4);
                }
            } else {
                #pragma unroll
                for (int i = 0; i < 4; ++i) {
                    int idx = tid + i * 256;
                    int r = idx >> 3, c4 = (idx & 7) << 2;
                    int gn = n0 + r;
                    int gnc = min(gn, Ndim - 1);
                    float4 v;
                    #pragma unroll
                    for (int j = 0; j < 4; ++j) {
                        int gk = k0 + c4 + j;
                        int gkc = min(gk, Kdim - 1);
                        float t = Bm[(long long)gnc * ldb + gkc];
                        (&v.x)[j] = (gn < Ndim && gk < Kdim) ? t : 0.0f;
                    }
                    rb[i] = v;
                }
            }
        } else {
            if (bColsFull && (k0 + BK <= Kdim)) {
                #pragma unroll
                for (int i = 0; i < 4; ++i) {
                    int idx = tid + i * 256;
                    int k = idx >> 5, n4 = (idx & 31) << 2;
                    rb[i] = *(const float4*)(Bm + (long long)(k0 + k) * ldb + n0 + n4);
                }
            } else {
                #pragma unroll
                for (int i = 0; i < 4; ++i) {
                    int idx = tid + i * 256;
                    int k = idx >> 5, n4 = (idx & 31) << 2;
                    int gk = k0 + k;
                    int gkc = min(gk, Kdim - 1);
                    float4 v;
                    #pragma unroll
                    for (int j = 0; j < 4; ++j) {
                        int gn = n0 + n4 + j;
                        int gnc = min(gn, Ndim - 1);
                        float t = Bm[(long long)gkc * ldb + gnc];
                        (&v.x)[j] = (gk < Kdim && gn < Ndim) ? t : 0.0f;
                    }
                    rb[i] = v;
                }
            }
        }
    };
    auto stageStore = [&](int buf) {
        unsigned short* pA = sA[buf];
        unsigned short* pB = sB[buf];
        #pragma unroll
        for (int i = 0; i < 4; ++i) {
            int idx = tid + i * 256;
            int r = idx >> 3, c4 = (idx & 7) << 2;
            u32x2 p;
            p.x = pack2bf(ra[i].x, ra[i].y);
            p.y = pack2bf(ra[i].z, ra[i].w);
            *(u32x2*)(pA + r * LDSK + c4) = p;
        }
        if (TRANSB) {
            #pragma unroll
            for (int i = 0; i < 4; ++i) {
                int idx = tid + i * 256;
                int r = idx >> 3, c4 = (idx & 7) << 2;
                u32x2 p;
                p.x = pack2bf(rb[i].x, rb[i].y);
                p.y = pack2bf(rb[i].z, rb[i].w);
                *(u32x2*)(pB + r * LDSK + c4) = p;
            }
        } else {
            #pragma unroll
            for (int i = 0; i < 4; ++i) {
                int idx = tid + i * 256;
                int k = idx >> 5, n4 = (idx & 31) << 2;
                #pragma unroll
                for (int j = 0; j < 4; ++j)
                    pB[(n4 + j) * LDSK + k] = f2bf((&rb[i].x)[j]);
            }
        }
    };
    auto compute = [&](int buf) {
        const unsigned short* pA = sA[buf];
        const unsigned short* pB = sB[buf];
        FragU bfrag[2];
        #pragma unroll
        for (int nt = 0; nt < 2; ++nt) {
            const unsigned short* p = pB + (wn * 32 + nt * 16 + rsel) * LDSK + hsel;
            bfrag[nt].u[0] = *(const u32x4*)p;
            bfrag[nt].u[1] = *(const u32x4*)(p + 16);
        }
        #pragma unroll
        for (int mt = 0; mt < 4; ++mt) {
            FragU af;
            const unsigned short* p = pA + (wm * 64 + mt * 16 + rsel) * LDSK + hsel;
            af.u[0] = *(const u32x4*)p;
            af.u[1] = *(const u32x4*)(p + 16);
            #pragma unroll
            for (int nt = 0; nt < 2; ++nt) {
                acc[mt][nt] = __builtin_amdgcn_wmma_f32_16x16x32_bf16(
                    false, af.bf, false, bfrag[nt].bf, (short)0, acc[mt][nt],
                    false, false);
            }
        }
    };

    // ---- software-pipelined main loop (double-buffered LDS) ----
    const int nk = (Kdim + BK - 1) / BK;
    fetchA(0);
    fetchB(0);
    stageStore(0);
    __syncthreads();
    for (int kt = 0; kt < nk; ++kt) {
        const bool more = (kt + 1) < nk;
        if (more) { fetchA((kt + 1) * BK); fetchB((kt + 1) * BK); }  // loads in flight
        compute(kt & 1);                                             // WMMA on current buffer
        if (more) stageStore((kt + 1) & 1);                          // convert + fill other buffer
        __syncthreads();
    }

    // ---- epilogue ----
    // acc VGPR r: lanes 0-15 -> M=r, lanes 16-31 -> M=r+8; N = lane%16
    const int mlane = (lane >> 4) * 8;
    const int nlane = lane & 15;
    float bcol[2];
    if (BIASMODE == 1) {
        #pragma unroll
        for (int nt = 0; nt < 2; ++nt) {
            int gn = min(n0 + wn * 32 + nt * 16 + nlane, Ndim - 1);
            bcol[nt] = biasz[gn];
        }
    }
    if ((m0 + BM <= Mdim) && (n0 + BN <= Ndim)) {
        #pragma unroll
        for (int mt = 0; mt < 4; ++mt)
            #pragma unroll
            for (int nt = 0; nt < 2; ++nt)
                #pragma unroll
                for (int r = 0; r < 8; ++r) {
                    int gm = m0 + wm * 64 + mt * 16 + r + mlane;
                    int gn = n0 + wn * 32 + nt * 16 + nlane;
                    float v = acc[mt][nt][r];
                    if (BIASMODE == 1) v += bcol[nt];
                    else if (BIASMODE == 2) v += biasz[gm];
                    if (GELU) v = 0.5f * v * (1.0f + erff(v * 0.70710678118654752f));
                    C[(long long)gm * ldc + gn] = v;
                }
    } else {
        #pragma unroll
        for (int mt = 0; mt < 4; ++mt)
            #pragma unroll
            for (int nt = 0; nt < 2; ++nt)
                #pragma unroll
                for (int r = 0; r < 8; ++r) {
                    int gm = m0 + wm * 64 + mt * 16 + r + mlane;
                    int gn = n0 + wn * 32 + nt * 16 + nlane;
                    if (gm < Mdim && gn < Ndim) {
                        float v = acc[mt][nt][r];
                        if (BIASMODE == 1) v += bcol[nt];
                        else if (BIASMODE == 2) v += biasz[gm];
                        if (GELU) v = 0.5f * v * (1.0f + erff(v * 0.70710678118654752f));
                        C[(long long)gm * ldc + gn] = v;
                    }
                }
    }
}

// ---------------- im2col ----------------
// patches[b, h*14+w, px*48+py*3+c] = images[b, c, px*14+h, py*14+w]
__global__ void im2col_kernel(const float* __restrict__ img, float* __restrict__ out,
                              long long total) {
    long long idx = (long long)blockIdx.x * blockDim.x + threadIdx.x;
    if (idx >= total) return;
    int d = (int)(idx % 768);
    long long rem = idx / 768;
    int p = (int)(rem % 196);
    int b = (int)(rem / 196);
    int h = p / 14, w = p % 14;
    int px = d / 48, py = (d % 48) / 3, c = d % 3;
    long long src = (((long long)b * Cc + c) * IMGc + (px * 14 + h)) * IMGc + (py * 14 + w);
    out[idx] = img[src];
}

// ---------------- assemble x0 = [cls; tokens] + pos_emb ----------------
__global__ void assemble_x0(const float* __restrict__ tok, const float* __restrict__ cls,
                            float* __restrict__ x, long long total) {
    long long idx = (long long)blockIdx.x * blockDim.x + threadIdx.x;
    if (idx >= total) return;
    int d = (int)(idx % Dc);
    long long rem = idx / Dc;
    int n = (int)(rem % Nc);
    int b = (int)(rem / Nc);
    float v;
    if (n == 0) v = cls[d];
    else        v = tok[((long long)b * 196 + (n - 1)) * Dc + d];
    int je = (d >> 1) << 1;
    float ang = (float)n / powf(10000.0f, (float)je / (float)Dc);
    float pe = (d & 1) ? cosf(ang) : sinf(ang);
    x[idx] = v + pe;
}

// ---------------- softmax over rows of length Nc (with scale) ----------------
__global__ __launch_bounds__(256) void softmax_rows(float* __restrict__ att, float scale) {
    __shared__ float red[256];
    long long row = blockIdx.x;
    float* p = att + row * Nc;
    int t = threadIdx.x;
    float v = (t < Nc) ? p[t] * scale : -3.0e38f;
    red[t] = v; __syncthreads();
    for (int s = 128; s > 0; s >>= 1) { if (t < s) red[t] = fmaxf(red[t], red[t + s]); __syncthreads(); }
    float mx = red[0]; __syncthreads();
    float e = (t < Nc) ? expf(v - mx) : 0.0f;
    red[t] = e; __syncthreads();
    for (int s = 128; s > 0; s >>= 1) { if (t < s) red[t] += red[t + s]; __syncthreads(); }
    float inv = 1.0f / red[0];
    if (t < Nc) p[t] = e * inv;
}

// ---------------- out = LayerNorm(X + Y) * g + b ----------------
__global__ __launch_bounds__(256)
void add_ln(const float* __restrict__ X, const float* __restrict__ Y,
            const float* __restrict__ g, const float* __restrict__ b,
            float* __restrict__ out) {
    __shared__ float red[256];
    long long row = blockIdx.x;
    const float* px = X + row * Dc;
    const float* py = Y + row * Dc;
    int t = threadIdx.x;
    float vloc[3];
    float s = 0.0f;
    #pragma unroll
    for (int i = 0; i < 3; ++i) { int d = t + i * 256; float v = px[d] + py[d]; vloc[i] = v; s += v; }
    red[t] = s; __syncthreads();
    for (int st = 128; st > 0; st >>= 1) { if (t < st) red[t] += red[t + st]; __syncthreads(); }
    float mu = red[0] * (1.0f / Dc); __syncthreads();
    float s2 = 0.0f;
    #pragma unroll
    for (int i = 0; i < 3; ++i) { float dd = vloc[i] - mu; s2 += dd * dd; }
    red[t] = s2; __syncthreads();
    for (int st = 128; st > 0; st >>= 1) { if (t < st) red[t] += red[t + st]; __syncthreads(); }
    float inv = rsqrtf(red[0] * (1.0f / Dc) + 1e-5f);
    #pragma unroll
    for (int i = 0; i < 3; ++i) {
        int d = t + i * 256;
        out[row * Dc + d] = (vloc[i] - mu) * inv * g[d] + b[d];
    }
}

// ---------------- copy out[:,1:,:] into a d_out segment ----------------
__global__ void copy_skip(const float* __restrict__ X, float* __restrict__ dst,
                          long long total) {
    long long idx = (long long)blockIdx.x * blockDim.x + threadIdx.x;
    if (idx >= total) return;
    int d = (int)(idx % Dc);
    long long rem = idx / Dc;
    int p = (int)(rem % 196);
    int b = (int)(rem / 196);
    dst[idx] = X[((long long)b * Nc + p + 1) * Dc + d];
}

// ---------------- host driver ----------------
extern "C" void kernel_launch(void* const* d_in, const int* in_sizes, int n_in,
                              void* d_out, int out_size, void* d_ws, size_t ws_size,
                              hipStream_t stream) {
    (void)in_sizes; (void)n_in; (void)out_size; (void)ws_size;
    const float* images = (const float*)d_in[0];
    const float* cls    = (const float*)d_in[1];
    const float* Wp     = (const float*)d_in[2];
    const float* bp     = (const float*)d_in[3];
    const float* Wq     = (const float*)d_in[4];
    const float* bq     = (const float*)d_in[5];
    const float* Wk     = (const float*)d_in[6];
    const float* bk     = (const float*)d_in[7];
    const float* Wv     = (const float*)d_in[8];
    const float* bv     = (const float*)d_in[9];
    const float* Wm     = (const float*)d_in[10];
    const float* bm     = (const float*)d_in[11];
    const float* g1     = (const float*)d_in[12];
    const float* be1    = (const float*)d_in[13];
    const float* W1     = (const float*)d_in[14];
    const float* b1     = (const float*)d_in[15];
    const float* W2     = (const float*)d_in[16];
    const float* b2     = (const float*)d_in[17];
    const float* g2     = (const float*)d_in[18];
    const float* be2    = (const float*)d_in[19];
    float* out = (float*)d_out;

    const long long BND  = (long long)Bc * Nc * Dc;          // 2,420,736
    const long long BHND = (long long)Bc * Hc * Nc * Dc;     // 29,048,832
    const long long BHNN = (long long)Bc * Hc * Nc * Nc;     // 7,451,328
    const long long SEG  = (long long)Bc * 196 * Dc;         // output segment

    float* ws  = (float*)d_ws;
    float* X   = ws;
    float* X1  = X   + BND;
    float* TMP = X1  + BND;
    float* Q   = TMP + BND;
    float* K   = Q   + BHND;
    float* V   = K   + BHND;
    float* O   = V   + BHND;
    float* ATT = O   + BHND;
    float* HDN = ATT + BHNN;
    float* PATCH = HDN;   // alias: only used pre-loop
    float* TOK   = TMP;   // alias: only used pre-loop

    const int BIG = 1 << 30;
    const int BNrows = Bc * Nc;   // 3152

    // ---- patch embedding ----
    {
        long long tot = (long long)Bc * 196 * 768;
        im2col_kernel<<<(unsigned)((tot + 255) / 256), 256, 0, stream>>>(images, PATCH, tot);
        dim3 g((768 + BN - 1) / BN, (Bc * 196 + BM - 1) / BM, 1);
        gemm_wmma<true, 1, false><<<g, 256, 0, stream>>>(
            PATCH, Wp, bp, TOK, Bc * 196, Dc, 768, 768, 768, Dc,
            0LL, 1, 0LL, 1, 0LL, 0LL);
        long long tot2 = BND;
        assemble_x0<<<(unsigned)((tot2 + 255) / 256), 256, 0, stream>>>(TOK, cls, X, tot2);
    }

    const dim3 gQKV((Dc + BN - 1) / BN, (Nc + BM - 1) / BM, Bc * Hc);  // 6 x 2 x 192
    const dim3 gATT((Nc + BN - 1) / BN, (Nc + BM - 1) / BM, Bc * Hc);  // 2 x 2 x 192
    const dim3 gMSA((Dc + BN - 1) / BN, (Nc + BM - 1) / BM, Bc);       // 6 x 2 x 16
    const dim3 gMLP1((Mc + BN - 1) / BN, (BNrows + BM - 1) / BM, 1);   // 24 x 25
    const dim3 gMLP2((Dc + BN - 1) / BN, (BNrows + BM - 1) / BM, 1);   // 6 x 25

    for (int l = 0; l < Lc; ++l) {
        const float* WqL = Wq + (long long)l * Hc * Dc * Dc;
        const float* WkL = Wk + (long long)l * Hc * Dc * Dc;
        const float* WvL = Wv + (long long)l * Hc * Dc * Dc;
        const float* bqL = bq + (long long)l * Hc * Dc;
        const float* bkL = bk + (long long)l * Hc * Dc;
        const float* bvL = bv + (long long)l * Hc * Dc;
        const float* WmL = Wm + (long long)l * Nc * (Hc * Nc);
        const float* bmL = bm + (long long)l * Nc;
        const float* W1L = W1 + (long long)l * Mc * Dc;
        const float* b1L = b1 + (long long)l * Mc;
        const float* W2L = W2 + (long long)l * Dc * Mc;
        const float* b2L = b2 + (long long)l * Dc;

        // Q/K/V : per (b,h): x[b] @ W[l,h]^T + bias[l,h]
        gemm_wmma<true, 1, false><<<gQKV, 256, 0, stream>>>(
            X, WqL, bqL, Q, Nc, Dc, Dc, Dc, Dc, Dc,
            (long long)Nc * Dc, Hc, (long long)Dc * Dc, Hc,
            (long long)Nc * Dc, (long long)Dc);
        gemm_wmma<true, 1, false><<<gQKV, 256, 0, stream>>>(
            X, WkL, bkL, K, Nc, Dc, Dc, Dc, Dc, Dc,
            (long long)Nc * Dc, Hc, (long long)Dc * Dc, Hc,
            (long long)Nc * Dc, (long long)Dc);
        gemm_wmma<true, 1, false><<<gQKV, 256, 0, stream>>>(
            X, WvL, bvL, V, Nc, Dc, Dc, Dc, Dc, Dc,
            (long long)Nc * Dc, Hc, (long long)Dc * Dc, Hc,
            (long long)Nc * Dc, (long long)Dc);

        // scores = Q @ K^T  (per b,h), then softmax with 1/sqrt(196)
        gemm_wmma<true, 0, false><<<gATT, 256, 0, stream>>>(
            Q, K, nullptr, ATT, Nc, Nc, Dc, Dc, Dc, Nc,
            (long long)Nc * Dc, 1, (long long)Nc * Dc, BIG,
            (long long)Nc * Nc, 0LL);
        softmax_rows<<<(unsigned)(Bc * Hc * Nc), 256, 0, stream>>>(ATT, 1.0f / 14.0f);

        // O = attn @ V (per b,h)
        gemm_wmma<false, 0, false><<<gQKV, 256, 0, stream>>>(
            ATT, V, nullptr, O, Nc, Dc, Nc, Nc, Dc, Dc,
            (long long)Nc * Nc, 1, (long long)Nc * Dc, BIG,
            (long long)Nc * Dc, 0LL);

        // MSA mix: msa[b] = Wm[l] (197 x 2364) @ O[b] (2364 x 768) + bm[l] (per-row)
        gemm_wmma<false, 2, false><<<gMSA, 256, 0, stream>>>(
            WmL, O, bmL, TMP, Nc, Dc, Hc * Nc, Hc * Nc, Dc, Dc,
            0LL, 1, (long long)Hc * Nc * Dc, BIG,
            (long long)Nc * Dc, 0LL);

        // x1 = LN(x + msa)
        add_ln<<<(unsigned)BNrows, 256, 0, stream>>>(X, TMP, g1 + (long long)l * Dc,
                                                     be1 + (long long)l * Dc, X1);

        // MLP
        gemm_wmma<true, 1, true><<<gMLP1, 256, 0, stream>>>(
            X1, W1L, b1L, HDN, BNrows, Mc, Dc, Dc, Dc, Mc,
            0LL, 1, 0LL, 1, 0LL, 0LL);
        gemm_wmma<true, 1, false><<<gMLP2, 256, 0, stream>>>(
            HDN, W2L, b2L, TMP, BNrows, Dc, Mc, Mc, Mc, Dc,
            0LL, 1, 0LL, 1, 0LL, 0LL);

        // x = LN(x1 + mlp)
        add_ln<<<(unsigned)BNrows, 256, 0, stream>>>(X1, TMP, g2 + (long long)l * Dc,
                                                     be2 + (long long)l * Dc, X);

        // skips after layers 3,6,9,12 -> segments 1..4
        if (l == 2 || l == 5 || l == 8 || l == 11) {
            int seg = (l + 1) / 3;
            copy_skip<<<(unsigned)((SEG + 255) / 256), 256, 0, stream>>>(
                X, out + (long long)seg * SEG, SEG);
        }
    }
    // final output -> segment 0
    copy_skip<<<(unsigned)((SEG + 255) / 256), 256, 0, stream>>>(X, out, SEG);
}
